// HRNN_84748294685333
// MI455X (gfx1250) — matrix operand order repaired and checked by reading, not compiled, not run on hardware
//
#include <hip/hip_runtime.h>

#define BB    32
#define TT    512
#define DD    768
#define HH    512
#define GG    1536
#define NWG   16
#define BLOCK 256
#define WPB   (BLOCK / 32)
#define GWAVES (NWG * WPB)

typedef __bf16 v16bf __attribute__((ext_vector_type(16)));
typedef float  v8f   __attribute__((ext_vector_type(8)));

union Frag16 { v16bf v; unsigned short s[16]; uint4 u[2]; };

__device__ __forceinline__ unsigned short f2bf(float f) {
  unsigned u = __float_as_uint(f);
  u += 0x7FFFu + ((u >> 16) & 1u);          // round-to-nearest-even
  return (unsigned short)(u >> 16);
}
__device__ __forceinline__ float bf2f(unsigned short s) {
  return __uint_as_float(((unsigned)s) << 16);
}
__device__ __forceinline__ float sigmoidf_(float x) {
  return 1.0f / (1.0f + __expf(-x));
}

// ---------------------------------------------------------------------------
// N-stripe WMMA worker: computes BOTH 16x16 C tiles of a 32-row x 16-col
// stripe (M=32 batch). The ENTIRE B stripe (16 fragments, 128 VGPRs) and the
// entire A panel (32 fragments, 256 VGPRs) are issued as one batch of b128
// loads before any WMMA, so all loads are in flight together and the compiler
// emits a descending s_wait_loadcnt ladder instead of a full stall per k-step.
// A layout (ISA 7.12.2, 16-bit A 16x32): lane holds two contiguous 8-elem
// chunks: k0..k0+7 and k0+16..k0+23, k0 = kt*32 + (lane>=16 ? 8 : 0).
// B fragments pre-swizzled: 1024B per (nt,kt) fragment, 32B per lane.
// ---------------------------------------------------------------------------
__device__ __forceinline__ void run_stripe(const unsigned short* __restrict__ A,
                                           const uint4* __restrict__ W,
                                           const float* __restrict__ bias,
                                           int nt,
                                           float* outf, unsigned short* outb,
                                           int Ncols, bool relu) {
  const int lane = threadIdx.x & 31;
  const float bv = bias[nt * 16 + (lane & 15)];
  const unsigned short* Ar0 = A + (size_t)(lane & 15) * HH;   // rows 0..15
  const unsigned short* Ar1 = Ar0 + (size_t)16 * HH;          // rows 16..31
  const int khalf = (lane & 16) ? 8 : 0;
  const uint4* wl = W + (size_t)nt * 16 * 64 + lane * 2;      // KT = 16

  Frag16 bf[16], a0[16], a1[16];
#pragma unroll
  for (int kt = 0; kt < 16; ++kt) {                           // whole B stripe
    bf[kt].u[0] = wl[kt * 64];
    bf[kt].u[1] = wl[kt * 64 + 1];
  }
#pragma unroll
  for (int kt = 0; kt < 16; ++kt) {                           // whole A panel
    const int k0 = (kt << 5) + khalf;
    a0[kt].u[0] = *(const uint4*)(Ar0 + k0);
    a0[kt].u[1] = *(const uint4*)(Ar0 + k0 + 16);
    a1[kt].u[0] = *(const uint4*)(Ar1 + k0);
    a1[kt].u[1] = *(const uint4*)(Ar1 + k0 + 16);
  }
  v8f acc0, acc1;
#pragma unroll
  for (int r = 0; r < 8; ++r) { acc0[r] = bv; acc1[r] = bv; }
#pragma unroll
  for (int kt = 0; kt < 16; ++kt) {
    acc0 = __builtin_amdgcn_wmma_f32_16x16x32_bf16(false, a0[kt].v, false, bf[kt].v,
                                                   (short)0, acc0, false, false);
    acc1 = __builtin_amdgcn_wmma_f32_16x16x32_bf16(false, a1[kt].v, false, bf[kt].v,
                                                   (short)0, acc1, false, false);
  }
  // C layout (ISA 7.12.2): VGPR r -> row r + (laneHi?8:0), col n = lane%16
  const int n = nt * 16 + (lane & 15);
  const int mb = (lane & 16) ? 8 : 0;
#pragma unroll
  for (int r = 0; r < 8; ++r) {
    float v0 = acc0[r], v1 = acc1[r];
    if (relu) { v0 = fmaxf(v0, 0.0f); v1 = fmaxf(v1, 0.0f); }
    if (outf) {
      outf[(size_t)(mb + r) * Ncols + n] = v0;
      outf[(size_t)(16 + mb + r) * Ncols + n] = v1;
    }
    if (outb) {
      outb[(size_t)(mb + r) * Ncols + n] = f2bf(v0);
      outb[(size_t)(16 + mb + r) * Ncols + n] = f2bf(v1);
    }
  }
}

// ---------------------------------------------------------------------------
// Parameter block for the persistent recurrent kernel
// ---------------------------------------------------------------------------
struct HP {
  const float *bhh0, *bih_l, *bhh_l, *mb1, *mb2, *mW3, *mb3;
  const unsigned short* gi0;
  const uint4 *Whh0sw, *Wihsw0, *Wihsw1, *Whhsw0, *Whhsw1;
  const uint4 *mW1sw0, *mW1sw1, *mW2sw0, *mW2sw1;
  float* h_f32;  unsigned short* h_bf16;
  float* c_f32;  unsigned short* c_bf16;
  float *gibuf, *ghbuf;
  unsigned short *m0a, *m1a;
  float *m0b, *m1b;
  float* out;
  unsigned *bcnt, *bgen;
};

__device__ __forceinline__ void grid_barrier(unsigned* cnt, unsigned* gen) {
  __syncthreads();
  if (threadIdx.x == 0) {
    __threadfence();
    volatile unsigned* vg = (volatile unsigned*)gen;
    unsigned g = *vg;
    unsigned old = atomicAdd(cnt, 1u);
    if (old == (unsigned)(NWG - 1)) {
      *(volatile unsigned*)cnt = 0u;
      __threadfence();
      *vg = g + 1u;
    } else {
      while (*vg == g) __builtin_amdgcn_s_sleep(1);
      __threadfence();
    }
  }
  __syncthreads();
}

// GRU combine: r=sig(ir+hr), z=sig(iz+hz), n=tanh(in + r*hn), c=(1-z)n+z*h
// gi already contains bih; gh already contains bhh (kept separate for n-gate).
__device__ __forceinline__ void gru_elem(int t, int l, int i, const HP& P) {
  const int b = i >> 9;
  const int j = i & (HH - 1);
  float ir, iz, in_;
  if (l == 0) {
    const unsigned short* g0 = P.gi0 + ((size_t)b * TT + t) * GG;
    ir = bf2f(g0[j]); iz = bf2f(g0[HH + j]); in_ = bf2f(g0[2 * HH + j]);
  } else {
    const float* gi = P.gibuf + (size_t)b * GG;
    ir = gi[j]; iz = gi[HH + j]; in_ = gi[2 * HH + j];
  }
  const float* gh = P.ghbuf + (size_t)b * GG;
  const float r = sigmoidf_(ir + gh[j]);
  const float z = sigmoidf_(iz + gh[HH + j]);
  const float n = tanhf(in_ + r * gh[2 * HH + j]);
  const float h = P.h_f32[(size_t)l * BB * HH + i];
  const float c = (1.0f - z) * n + z * h;
  P.c_f32[(size_t)l * BB * HH + i] = c;
  P.c_bf16[(size_t)l * BB * HH + i] = f2bf(c);
}

// ---------------------------------------------------------------------------
// Persistent recurrent kernel: 16 WGs x 8 wave32, 7 phases per timestep.
// ---------------------------------------------------------------------------
__global__ void __launch_bounds__(BLOCK) hrnn_persistent(HP P) {
  const int gw   = blockIdx.x * WPB + (threadIdx.x >> 5);
  const int gtid = blockIdx.x * BLOCK + threadIdx.x;
  const int lane = threadIdx.x & 31;
  const int widx = threadIdx.x >> 5;
  __shared__ float s_p[2][BB];

  for (int t = 0; t < TT; ++t) {
    // P1: gh0 = h0 @ Whh0^T + bhh0     (96 N-stripes)
    for (int id = gw; id < 96; id += GWAVES)
      run_stripe(P.h_bf16, P.Whh0sw, P.bhh0, id, P.ghbuf,
                 (unsigned short*)0, GG, false);
    grid_barrier(P.bcnt, P.bgen);

    // P2: GRU layer0 -> c0 (f32 + bf16)
    for (int i = gtid; i < BB * HH; i += NWG * BLOCK) gru_elem(t, 0, i, P);
    grid_barrier(P.bcnt, P.bgen);

    // P3: gi1 = c0@Wih1^T, gh1 = h1@Whh1^T, m0a = relu(c0@mW1_0+b)  (224)
    for (int id = gw; id < 224; id += GWAVES) {
      if (id < 96)
        run_stripe(P.c_bf16, P.Wihsw0, P.bih_l, id, P.gibuf,
                   (unsigned short*)0, GG, false);
      else if (id < 192)
        run_stripe(P.h_bf16 + BB * HH, P.Whhsw0, P.bhh_l, id - 96, P.ghbuf,
                   (unsigned short*)0, GG, false);
      else
        run_stripe(P.c_bf16, P.mW1sw0, P.mb1, id - 192, (float*)0, P.m0a,
                   HH, true);
    }
    grid_barrier(P.bcnt, P.bgen);

    // P4: GRU layer1 -> c1
    for (int i = gtid; i < BB * HH; i += NWG * BLOCK) gru_elem(t, 1, i, P);
    grid_barrier(P.bcnt, P.bgen);

    // P5: gi2, gh2, m1a = relu(c1@mW1_1), m0b = relu(m0a@mW2_0)   (256)
    for (int id = gw; id < 256; id += GWAVES) {
      if (id < 96)
        run_stripe(P.c_bf16 + BB * HH, P.Wihsw1, P.bih_l + GG, id,
                   P.gibuf, (unsigned short*)0, GG, false);
      else if (id < 192)
        run_stripe(P.h_bf16 + 2 * BB * HH, P.Whhsw1, P.bhh_l + GG, id - 96,
                   P.ghbuf, (unsigned short*)0, GG, false);
      else if (id < 224)
        run_stripe(P.c_bf16 + BB * HH, P.mW1sw1, P.mb1 + HH, id - 192,
                   (float*)0, P.m1a, HH, true);
      else
        run_stripe(P.m0a, P.mW2sw0, P.mb2, id - 224, P.m0b,
                   (unsigned short*)0, HH, true);
    }
    grid_barrier(P.bcnt, P.bgen);

    // P6: m1b = relu(m1a@mW2_1)  (WGs 0-3, 32 stripes) || GRU2 (WGs 4-15)
    if (blockIdx.x < 4) {
      const int id = blockIdx.x * WPB + widx;        // 0..31, 1 stripe/wave
      run_stripe(P.m1a, P.mW2sw1, P.mb2 + HH, id, P.m1b,
                 (unsigned short*)0, HH, true);
    } else {
      for (int i = (blockIdx.x - 4) * BLOCK + threadIdx.x; i < BB * HH;
           i += 12 * BLOCK)
        gru_elem(t, 2, i, P);
    }
    grid_barrier(P.bcnt, P.bgen);

    // P7: boundary probs (redundant per WG, wave-shuffle reduction) + mixing
    for (int task = widx; task < 2 * BB; task += WPB) {
      const int prob = task & 1;
      const int b = task >> 1;
      const float* src = (prob ? P.m1b : P.m0b) + (size_t)b * HH;
      const float* w = P.mW3 + (size_t)prob * HH;
      float s = 0.0f;
      for (int j = lane; j < HH; j += 32) s += src[j] * w[j];
#pragma unroll
      for (int o = 16; o > 0; o >>= 1) s += __shfl_xor(s, o, 32);
      if (lane == 0) s_p[prob][b] = sigmoidf_(s + P.mb3[prob]);
    }
    __syncthreads();
    for (int i = gtid; i < BB * HH; i += NWG * BLOCK) {
      const int b = i >> 9;
      const int j = i & (HH - 1);
      const float p0 = s_p[0][b], p1 = s_p[1][b];
      const float c0 = P.c_f32[i];
      const float c1 = P.c_f32[BB * HH + i];
      const float c2 = P.c_f32[2 * BB * HH + i];
      // hdx=0: softmax over {1-p0, p0(1-p1), p0*p1}
      float a0 = 1.0f - p0, a1 = p0 * (1.0f - p1), a2 = p0 * p1;
      float mx = fmaxf(a0, fmaxf(a1, a2));
      float e0 = __expf(a0 - mx), e1 = __expf(a1 - mx), e2 = __expf(a2 - mx);
      float inv = 1.0f / (e0 + e1 + e2);
      const float h0 = (e0 * c0 + e1 * c1 + e2 * c2) * inv;
      // hdx=1: softmax over {1-p1, p1}
      float b0 = 1.0f - p1, b1v = p1;
      float m2 = fmaxf(b0, b1v);
      float f0 = __expf(b0 - m2), f1 = __expf(b1v - m2);
      float inv2 = 1.0f / (f0 + f1);
      const float h1 = (f0 * c1 + f1 * c2) * inv2;
      const float h2 = c2;
      P.h_f32[i] = h0;                P.h_bf16[i] = f2bf(h0);
      P.h_f32[BB * HH + i] = h1;      P.h_bf16[BB * HH + i] = f2bf(h1);
      P.h_f32[2 * BB * HH + i] = h2;  P.h_bf16[2 * BB * HH + i] = f2bf(h2);
      if (j == 0) {
        float* o = P.out + ((size_t)b * TT + t) * 2;
        o[0] = p0; o[1] = p1;
      }
    }
    grid_barrier(P.bcnt, P.bgen);
  }
}

// ---------------------------------------------------------------------------
// Precompute gi0[b,t,:] = x[b,t,:] @ Wih0^T + bih0 (bf16 out).
// Stripe = 32 consecutive timesteps of one batch row x one 16-col N-tile,
// sharing B fragments between the two WMMAs per k-step. Massively parallel,
// latency hidden by occupancy.
// ---------------------------------------------------------------------------
__global__ void __launch_bounds__(BLOCK) gi0_kernel(const float* __restrict__ x,
                                                    const uint4* __restrict__ Wsw,
                                                    const float* __restrict__ bih0,
                                                    unsigned short* __restrict__ gi0) {
  const int wid = blockIdx.x * WPB + (threadIdx.x >> 5);   // 0..49151
  const int lane = threadIdx.x & 31;
  const int nt = wid % (GG / 16);
  const int btt = wid / (GG / 16);                          // 0..511
  const int b = btt >> 4;                                   // TT/32 = 16 groups
  const int t0 = (btt & 15) << 5;
  const float* xr0 = x + ((size_t)b * TT + t0 + (lane & 15)) * DD;
  const float* xr1 = xr0 + (size_t)16 * DD;
  const int khalf = (lane & 16) ? 8 : 0;
  const int KT = DD >> 5;                                   // 24
  v8f acc0, acc1;
#pragma unroll
  for (int r = 0; r < 8; ++r) { acc0[r] = 0.0f; acc1[r] = 0.0f; }
  const uint4* wl = Wsw + (size_t)nt * KT * 64 + lane * 2;
  for (int kt = 0; kt < KT; ++kt) {
    Frag16 a0, a1, bf;
    const int k0 = (kt << 5) + khalf;
#pragma unroll
    for (int e = 0; e < 8; ++e) {
      a0.s[e]     = f2bf(xr0[k0 + e]);
      a0.s[8 + e] = f2bf(xr0[k0 + 16 + e]);
      a1.s[e]     = f2bf(xr1[k0 + e]);
      a1.s[8 + e] = f2bf(xr1[k0 + 16 + e]);
    }
    bf.u[0] = wl[0]; bf.u[1] = wl[1]; wl += 64;
    acc0 = __builtin_amdgcn_wmma_f32_16x16x32_bf16(false, a0.v, false, bf.v,
                                                   (short)0, acc0, false, false);
    acc1 = __builtin_amdgcn_wmma_f32_16x16x32_bf16(false, a1.v, false, bf.v,
                                                   (short)0, acc1, false, false);
  }
  const int n = nt * 16 + (lane & 15);
  const float bias = bih0[n];
  const int mbase = t0 + ((lane & 16) ? 8 : 0);
#pragma unroll
  for (int r = 0; r < 8; ++r) {
    gi0[((size_t)b * TT + mbase + r) * GG + n]      = f2bf(acc0[r] + bias);
    gi0[((size_t)b * TT + 16 + mbase + r) * GG + n] = f2bf(acc1[r] + bias);
  }
}

// Convert one (N,K) fp32 weight into bf16 WMMA-B fragments:
// dst[((nt*KT+kt)*32+lane)*16+e] = W[nt*16 + lane%16][kt*32 + (lane>=16?8:0) + (e&7) + (e>=8?16:0)]
__global__ void __launch_bounds__(BLOCK) wconv_kernel(const float* __restrict__ src,
                                                      unsigned short* __restrict__ dst,
                                                      int N, int K) {
  const int KT = K >> 5;
  const size_t total = (size_t)N * K;
  for (size_t d = (size_t)blockIdx.x * blockDim.x + threadIdx.x; d < total;
       d += (size_t)gridDim.x * blockDim.x) {
    const int e = (int)(d & 15);
    const int lane = (int)((d >> 4) & 31);
    const size_t fk = d >> 9;
    const int kt = (int)(fk % KT);
    const int nt = (int)(fk / KT);
    const int n = nt * 16 + (lane & 15);
    const int k = (kt << 5) + ((lane & 16) ? 8 : 0) + (e & 7) + ((e & 8) ? 16 : 0);
    dst[d] = f2bf(src[(size_t)n * K + k]);
  }
}

__global__ void __launch_bounds__(BLOCK) init_kernel(float* h_f32,
                                                     unsigned short* h_bf16,
                                                     unsigned* bar) {
  const size_t i = (size_t)blockIdx.x * blockDim.x + threadIdx.x;
  if (i < 8) bar[i] = 0u;
  for (size_t k = i; k < (size_t)3 * BB * HH; k += (size_t)gridDim.x * blockDim.x) {
    h_f32[k] = 0.0f;
    h_bf16[k] = 0;
  }
}

// ---------------------------------------------------------------------------
extern "C" void kernel_launch(void* const* d_in, const int* in_sizes, int n_in,
                              void* d_out, int out_size, void* d_ws, size_t ws_size,
                              hipStream_t stream) {
  (void)in_sizes; (void)n_in; (void)out_size; (void)ws_size;
  const float* x    = (const float*)d_in[0];
  const float* Wih0 = (const float*)d_in[1];
  const float* Whh0 = (const float*)d_in[2];
  const float* bih0 = (const float*)d_in[3];
  const float* bhh0 = (const float*)d_in[4];
  const float* Wih  = (const float*)d_in[5];
  const float* Whh  = (const float*)d_in[6];
  const float* bih  = (const float*)d_in[7];
  const float* bhh  = (const float*)d_in[8];
  const float* mW1  = (const float*)d_in[9];
  const float* mb1  = (const float*)d_in[10];
  const float* mW2  = (const float*)d_in[11];
  const float* mb2  = (const float*)d_in[12];
  const float* mW3  = (const float*)d_in[13];
  const float* mb3  = (const float*)d_in[14];

  char* ws = (char*)d_ws;
  size_t off = 0;
  auto alloc = [&](size_t bytes) -> char* {
    char* p = ws + off;
    off = (off + bytes + 255) & ~(size_t)255;
    return p;
  };
  unsigned*       bar     = (unsigned*)alloc(256);
  unsigned short* gi0     = (unsigned short*)alloc((size_t)BB * TT * GG * 2);  // 50.3 MB
  unsigned short* Wih0sw  = (unsigned short*)alloc((size_t)GG * DD * 2);
  unsigned short* Whh0sw  = (unsigned short*)alloc((size_t)GG * HH * 2);
  unsigned short* Wihsw   = (unsigned short*)alloc((size_t)2 * GG * HH * 2);
  unsigned short* Whhsw   = (unsigned short*)alloc((size_t)2 * GG * HH * 2);
  unsigned short* mW1sw   = (unsigned short*)alloc((size_t)2 * HH * HH * 2);
  unsigned short* mW2sw   = (unsigned short*)alloc((size_t)2 * HH * HH * 2);
  float*          h_f32   = (float*)alloc((size_t)3 * BB * HH * 4);
  unsigned short* h_bf16  = (unsigned short*)alloc((size_t)3 * BB * HH * 2);
  float*          c_f32   = (float*)alloc((size_t)3 * BB * HH * 4);
  unsigned short* c_bf16  = (unsigned short*)alloc((size_t)3 * BB * HH * 2);
  float*          gibuf   = (float*)alloc((size_t)BB * GG * 4);
  float*          ghbuf   = (float*)alloc((size_t)BB * GG * 4);
  unsigned short* m0a     = (unsigned short*)alloc((size_t)BB * HH * 2);
  unsigned short* m1a     = (unsigned short*)alloc((size_t)BB * HH * 2);
  float*          m0b     = (float*)alloc((size_t)BB * HH * 4);
  float*          m1b     = (float*)alloc((size_t)BB * HH * 4);
  // total ~64 MB of workspace

  init_kernel<<<64, BLOCK, 0, stream>>>(h_f32, h_bf16, bar);

  wconv_kernel<<<1024, BLOCK, 0, stream>>>(Wih0, Wih0sw, GG, DD);
  wconv_kernel<<<1024, BLOCK, 0, stream>>>(Whh0, Whh0sw, GG, HH);
  wconv_kernel<<<1024, BLOCK, 0, stream>>>(Wih, Wihsw, GG, HH);
  wconv_kernel<<<1024, BLOCK, 0, stream>>>(Wih + (size_t)GG * HH, Wihsw + (size_t)GG * HH, GG, HH);
  wconv_kernel<<<1024, BLOCK, 0, stream>>>(Whh, Whhsw, GG, HH);
  wconv_kernel<<<1024, BLOCK, 0, stream>>>(Whh + (size_t)GG * HH, Whhsw + (size_t)GG * HH, GG, HH);
  wconv_kernel<<<1024, BLOCK, 0, stream>>>(mW1, mW1sw, HH, HH);
  wconv_kernel<<<1024, BLOCK, 0, stream>>>(mW1 + (size_t)HH * HH, mW1sw + (size_t)HH * HH, HH, HH);
  wconv_kernel<<<1024, BLOCK, 0, stream>>>(mW2, mW2sw, HH, HH);
  wconv_kernel<<<1024, BLOCK, 0, stream>>>(mW2 + (size_t)HH * HH, mW2sw + (size_t)HH * HH, HH, HH);

  // (B*T/32) * (G/16) = 49152 stripes, 8 waves/block
  gi0_kernel<<<(BB * TT / 32) * (GG / 16) / WPB, BLOCK, 0, stream>>>(
      x, (const uint4*)Wih0sw, bih0, gi0);

  HP P;
  P.bhh0 = bhh0; P.bih_l = bih; P.bhh_l = bhh;
  P.mb1 = mb1; P.mb2 = mb2; P.mW3 = mW3; P.mb3 = mb3;
  P.gi0 = gi0;
  P.Whh0sw = (const uint4*)Whh0sw;
  P.Wihsw0 = (const uint4*)Wihsw;
  P.Wihsw1 = (const uint4*)(Wihsw + (size_t)GG * HH);
  P.Whhsw0 = (const uint4*)Whhsw;
  P.Whhsw1 = (const uint4*)(Whhsw + (size_t)GG * HH);
  P.mW1sw0 = (const uint4*)mW1sw;
  P.mW1sw1 = (const uint4*)(mW1sw + (size_t)HH * HH);
  P.mW2sw0 = (const uint4*)mW2sw;
  P.mW2sw1 = (const uint4*)(mW2sw + (size_t)HH * HH);
  P.h_f32 = h_f32; P.h_bf16 = h_bf16;
  P.c_f32 = c_f32; P.c_bf16 = c_bf16;
  P.gibuf = gibuf; P.ghbuf = ghbuf;
  P.m0a = m0a; P.m1a = m1a; P.m0b = m0b; P.m1b = m1b;
  P.out = (float*)d_out;
  P.bcnt = bar; P.bgen = bar + 1;

  hrnn_persistent<<<NWG, BLOCK, 0, stream>>>(P);
}
